// DeformableTemporalConv1D_8074538516825
// MI455X (gfx1250) — matrix operand order repaired and checked by reading, not compiled, not run on hardware
//
#include <hip/hip_runtime.h>

typedef __attribute__((ext_vector_type(16))) _Float16 v16h;
typedef __attribute__((ext_vector_type(8)))  _Float16 v8h;
typedef __attribute__((ext_vector_type(4)))  _Float16 v4h;
typedef __attribute__((ext_vector_type(8)))  float    v8f;
typedef __attribute__((ext_vector_type(4)))  unsigned int v4u;
typedef __attribute__((ext_vector_type(8)))  int      v8i;
typedef __attribute__((ext_vector_type(4)))  int      v4i;

#define B_  4
#define C_  128
#define H_  512
#define W_  128
#define KT  3

// ---- Tensor Data Mover availability / arity selection ----
#if defined(__has_builtin)
#if __has_builtin(__builtin_amdgcn_tensor_load_to_lds)
#define HAVE_TDM 1
#endif
#endif
#if __has_include(<hip/amd_detail/amd_gfx1250_TDM.h>)
#define HAVE_TDM6 1   // therock headers -> 6-arg builtin
#endif

// ---- LDS layout (dynamic). Rows padded to break bank conflicts. ----
#define SAMP_STRIDE 136                               // halves per row (272 B = 68 dwords)
#define SAMP_BYTES  (KT * W_ * SAMP_STRIDE * 2)      // 104448
#define XROW_STRIDE 136
#define XROW_BYTES  (130 * XROW_STRIDE * 2)          // 35360 (overlaid on samp)
#define OFFW_STRIDE 392                               // halves per row (784 B = 196 dwords)
#define OFFW_BYTES  (16 * OFFW_STRIDE * 2)           // 12544
#define LDS_OFFW_OFF XROW_BYTES                       // 35360 (inside samp region)
#define LDS_OFFM_OFF SAMP_BYTES                       // 104448
#define LDS_TOTAL   (LDS_OFFM_OFF + 9 * W_ * 4)      // 109056

// ---------------------------------------------------------------------------
// Kernel 1: NCHW -> NHWC transpose of x into workspace (coalesced both sides)
// ---------------------------------------------------------------------------
__global__ __launch_bounds__(256)
void transpose_nchw_nhwc(const float* __restrict__ x, float* __restrict__ xt) {
    __shared__ float tile[32][33];
    const int blk    = blockIdx.x;        // (b*H_ + h)*16 + tileid
    const int tileid = blk & 15;
    const int bh     = blk >> 4;
    const int h      = bh & (H_ - 1);
    const int b      = bh >> 9;           // H_ = 512
    const int c0 = (tileid >> 2) * 32;
    const int w0 = (tileid & 3) * 32;
    for (int idx = threadIdx.x; idx < 32 * 32; idx += 256) {
        const int r = idx >> 5, col = idx & 31;
        tile[r][col] = x[(((size_t)b * C_ + (c0 + r)) * H_ + h) * W_ + (w0 + col)];
    }
    __syncthreads();
    for (int idx = threadIdx.x; idx < 32 * 32; idx += 256) {
        const int r = idx >> 5, col = idx & 31;
        xt[(((size_t)b * H_ + h) * W_ + (w0 + r)) * C_ + (c0 + col)] = tile[col][r];
    }
}

// ---------------------------------------------------------------------------
// Kernel 2: pack conv_w (O,C,1,K) fp32 -> f16 [j][o][c] (WMMA-B friendly)
// ---------------------------------------------------------------------------
__global__ __launch_bounds__(256)
void pack_weights(const float* __restrict__ conv_w, _Float16* __restrict__ wt16) {
    const int i = blockIdx.x * 256 + threadIdx.x;   // j*C_*C_ + o*C_ + c
    if (i >= KT * C_ * C_) return;
    const int c = i & (C_ - 1);
    const int o = (i >> 7) & (C_ - 1);
    const int j = i >> 14;
    wt16[i] = (_Float16)conv_w[(o * C_ + c) * KT + j];
}

// ---------------------------------------------------------------------------
// Kernel 3: pack offset+mask conv weights fp32 -> f16 [16][OFFW_STRIDE]
//           row ch (0..5 = offsets, 6..8 = mask, 9..15 = zero), k = j*128+c
// ---------------------------------------------------------------------------
__global__ __launch_bounds__(256)
void pack_offw(const float* __restrict__ off_w, const float* __restrict__ mask_w,
               _Float16* __restrict__ ow16) {
    const int i = blockIdx.x * 256 + threadIdx.x;
    if (i >= 16 * OFFW_STRIDE) return;
    const int ch = i / OFFW_STRIDE;
    const int k  = i - ch * OFFW_STRIDE;
    float v = 0.0f;
    if (k < KT * C_ && ch < 9) {
        const int j = k >> 7;
        const int c = k & (C_ - 1);
        v = (ch < 6) ? off_w[(ch * C_ + c) * KT + j]
                     : mask_w[((ch - 6) * C_ + c) * KT + j];
    }
    ow16[i] = (_Float16)v;
}

// ---------------------------------------------------------------------------
// TDM: 1-row tile of n dwords, ws -> LDS. Descriptor per CDNA5 ISA D# layout.
// ---------------------------------------------------------------------------
#if defined(HAVE_TDM)
__device__ __forceinline__ void tdm_load_1d(unsigned int lds_off, const void* gptr,
                                            unsigned int ndwords) {
    const unsigned long long ga = (unsigned long long)(uintptr_t)gptr;
    v4u g0;
    g0[0] = 1u;                                        // count=1, user mode
    g0[1] = lds_off;                                   // lds_addr (bytes)
    g0[2] = (unsigned int)(ga & 0xFFFFFFFFu);          // global_addr[31:0]
    g0[3] = (unsigned int)((ga >> 32) & 0x1FFFFFFu)    // global_addr[56:32]
          | (2u << 30);                                // type = 2 ("image")
    v8i g1;
    g1[0] = (int)(2u << 16);                           // data_size = 4 bytes
    g1[1] = (int)((ndwords & 0xFFFFu) << 16);          // tensor_dim0[15:0] @ bits 63:48
    g1[2] = (int)((ndwords >> 16) | (1u << 16));       // tensor_dim0[31:16] | tensor_dim1=1
    g1[3] = (int)((ndwords & 0xFFFFu) << 16);          // tile_dim0 @ bits 127:112
    g1[4] = 1;                                         // tile_dim1 = 1
    g1[5] = (int)ndwords;                              // tensor_dim0_stride[31:0]
    g1[6] = 0;
    g1[7] = 0;
    const v4i z4 = {0, 0, 0, 0};
#if defined(HAVE_TDM6)
    const v8i z8 = {0, 0, 0, 0, 0, 0, 0, 0};
    __builtin_amdgcn_tensor_load_to_lds(g0, g1, z4, z4, z8, 0);
#else
    __builtin_amdgcn_tensor_load_to_lds(g0, g1, z4, z4, 0);
#endif
}
#endif

// ---------------------------------------------------------------------------
// Main kernel: one block per (b,h) row. 256 threads = 8 waves.
// ---------------------------------------------------------------------------
__global__ __launch_bounds__(256)
void deform_conv_main(const float* __restrict__ xt,
                      const _Float16* __restrict__ wt16g,
                      const _Float16* __restrict__ offw16g,
                      const float* __restrict__ off_b,
                      const float* __restrict__ mask_b,
                      const float* __restrict__ conv_b,
                      float* __restrict__ out) {
    extern __shared__ char smem[];
    _Float16* samp    = (_Float16*)(smem);                 // [KT][W_][SAMP_STRIDE]
    _Float16* xrowh   = (_Float16*)(smem);                 // [130][XROW_STRIDE], overlaid
    _Float16* offw    = (_Float16*)(smem + LDS_OFFW_OFF);  // [16][OFFW_STRIDE]
    float*    offm    = (float*)   (smem + LDS_OFFM_OFF);  // [9][W_]

    const int tid = threadIdx.x;
    const int bh  = blockIdx.x;
    const int h   = bh & (H_ - 1);
    const int b   = bh >> 9;

    const int wave  = tid >> 5;
    const int lane  = tid & 31;
    const int hi    = lane >> 4;       // K-half selector
    const int ln    = lane & 15;       // M (A) / N (B,C,D) index
    const int mbase = wave * 16;

    // ---- Phase 0: kick TDM fetch of offset/mask weight block into LDS ----
#if defined(HAVE_TDM)
    if (tid == 0) {
        tdm_load_1d(LDS_OFFW_OFF, (const void*)offw16g, OFFW_BYTES / 4);
    }
#endif

    // ---- Phase 1a: stage x row (positions -1..128, zero-padded) as f16 ----
    {
        const float* xrg = xt + ((size_t)(b * H_ + h)) * W_ * C_;
        for (int i = tid; i < 130 * C_; i += 256) {
            const int c = i & (C_ - 1);
            const int p = i >> 7;              // 0..129 -> w = p-1
            const int w = p - 1;
            const float v = ((unsigned)w < W_) ? xrg[(size_t)w * C_ + c] : 0.0f;
            xrowh[p * XROW_STRIDE + c] = (_Float16)v;
        }
    }
#if defined(HAVE_TDM)
    if (tid < 32) __builtin_amdgcn_s_wait_tensorcnt(0);
#else
    {
        const unsigned int* src = (const unsigned int*)offw16g;
        unsigned int* dst = (unsigned int*)offw;
        for (int i = tid; i < OFFW_BYTES / 4; i += 256) dst[i] = src[i];
    }
#endif
    __syncthreads();

    // ---- Phase 1b: offset/mask conv on WMMA. M=pos tile, N=9(16) ch, K=384 ----
    // A[pos][j*128+c] = xrowh[pos+j][c]  (row-shifted views, no expansion)
    {
        v8f acc = {};
        #pragma unroll
        for (int j = 0; j < KT; ++j) {
            const _Float16* abase = xrowh + (mbase + ln + j) * XROW_STRIDE;
            const _Float16* bbase = offw + ln * OFFW_STRIDE + j * C_;
            #pragma unroll
            for (int kc = 0; kc < C_; kc += 32) {
                const _Float16* ap = abase + kc + hi * 8;
                const v8h alo = *(const v8h*)(ap);
                const v8h ahi = *(const v8h*)(ap + 16);
                const v16h a = __builtin_shufflevector(alo, ahi,
                        0, 1, 2, 3, 4, 5, 6, 7, 8, 9, 10, 11, 12, 13, 14, 15);
                const _Float16* bp = bbase + kc + hi * 8;
                const v8h blo = *(const v8h*)(bp);
                const v8h bhi = *(const v8h*)(bp + 16);
                const v16h bfrag = __builtin_shufflevector(blo, bhi,
                        0, 1, 2, 3, 4, 5, 6, 7, 8, 9, 10, 11, 12, 13, 14, 15);
                acc = __builtin_amdgcn_wmma_f32_16x16x32_f16(
                        false, a, false, bfrag, (short)0, acc, false, false);
            }
        }
        const int ch = ln;                   // C/D: lane -> N = channel
        if (ch < 9) {
            const float bias = (ch < 6) ? off_b[ch] : mask_b[ch - 6];
            #pragma unroll
            for (int r = 0; r < 8; ++r) {    // VGPR r -> M = pos = mbase + r + 8*hi
                float v = acc[r] + bias;
                if (ch >= 6) v = 1.0f / (1.0f + __expf(-v));   // sigmoid
                offm[ch * W_ + mbase + r + 8 * hi] = v;
            }
        }
    }
    __syncthreads();

    // ---- Phase 2: bilinear sampling -> samp f16 [j][pos][.] (mask folded in) ----
    for (int si = tid; si < KT * W_ * 2; si += 256) {
        const int j   = si >> 8;               // 256 items per tap
        const int rem = si & 255;
        const int pos = rem >> 1;
        const int c0  = (rem & 1) << 6;        // 0 or 64
        const float dy = offm[(2 * j)     * W_ + pos];
        const float dx = offm[(2 * j + 1) * W_ + pos];
        const float m  = offm[(6 + j)     * W_ + pos];
        const float y  = (float)h + dy;
        const float xc = (float)(pos - 1 + j) + dx;
        const float y0 = floorf(y), x0 = floorf(xc);
        const float fy = y - y0,    fx = xc - x0;
        const int iy0 = (int)y0, ix0 = (int)x0;
        float w00 = (1.0f - fy) * (1.0f - fx);
        float w01 = (1.0f - fy) * fx;
        float w10 = fy * (1.0f - fx);
        float w11 = fy * fx;
        const bool vy0 = (unsigned)iy0       < H_;
        const bool vy1 = (unsigned)(iy0 + 1) < H_;
        const bool vx0 = (unsigned)ix0       < W_;
        const bool vx1 = (unsigned)(ix0 + 1) < W_;
        if (!(vy0 && vx0)) w00 = 0.0f;
        if (!(vy0 && vx1)) w01 = 0.0f;
        if (!(vy1 && vx0)) w10 = 0.0f;
        if (!(vy1 && vx1)) w11 = 0.0f;
        const int cy0 = min(max(iy0, 0), H_ - 1);
        const int cy1 = min(max(iy0 + 1, 0), H_ - 1);
        const int cx0 = min(max(ix0, 0), W_ - 1);
        const int cx1 = min(max(ix0 + 1, 0), W_ - 1);
        const float4* q00 = (const float4*)(xt + (((size_t)b * H_ + cy0) * W_ + cx0) * C_ + c0);
        const float4* q01 = (const float4*)(xt + (((size_t)b * H_ + cy0) * W_ + cx1) * C_ + c0);
        const float4* q10 = (const float4*)(xt + (((size_t)b * H_ + cy1) * W_ + cx0) * C_ + c0);
        const float4* q11 = (const float4*)(xt + (((size_t)b * H_ + cy1) * W_ + cx1) * C_ + c0);
        _Float16* sdst = samp + (j * W_ + pos) * SAMP_STRIDE + c0;
        #pragma unroll 4
        for (int c4 = 0; c4 < 16; ++c4) {
            const float4 a0 = q00[c4], a1 = q01[c4], a2 = q10[c4], a3 = q11[c4];
            const float v0 = w00 * a0.x + w01 * a1.x + w10 * a2.x + w11 * a3.x;
            const float v1 = w00 * a0.y + w01 * a1.y + w10 * a2.y + w11 * a3.y;
            const float v2 = w00 * a0.z + w01 * a1.z + w10 * a2.z + w11 * a3.z;
            const float v3 = w00 * a0.w + w01 * a1.w + w10 * a2.w + w11 * a3.w;
            const v4h s = { (_Float16)(v0 * m), (_Float16)(v1 * m),
                            (_Float16)(v2 * m), (_Float16)(v3 * m) };
            *(v4h*)(sdst + 4 * c4) = s;
        }
    }
    __syncthreads();

    // ---- Phase 3: main WMMA GEMM. M=pos (16/wave), N=o (8 tiles), K=3x128 ----
    for (int nt = 0; nt < 8; ++nt) {
        const int nbase = nt * 16;
        v8f acc = {};
        for (int j = 0; j < KT; ++j) {
            const _Float16* abase = samp + (j * W_ + (mbase + ln)) * SAMP_STRIDE;
            const _Float16* bbase = wt16g + ((size_t)(j * C_ + (nbase + ln))) * C_;
            #pragma unroll
            for (int kc = 0; kc < C_; kc += 32) {
                const _Float16* ap = abase + kc + hi * 8;
                const v8h alo = *(const v8h*)(ap);
                const v8h ahi = *(const v8h*)(ap + 16);
                const v16h a = __builtin_shufflevector(alo, ahi,
                        0, 1, 2, 3, 4, 5, 6, 7, 8, 9, 10, 11, 12, 13, 14, 15);
                const v8h blo = *(const v8h*)(bbase + kc + hi * 8);
                const v8h bhi = *(const v8h*)(bbase + kc + hi * 8 + 16);
                const v16h bfrag = __builtin_shufflevector(blo, bhi,
                        0, 1, 2, 3, 4, 5, 6, 7, 8, 9, 10, 11, 12, 13, 14, 15);
                acc = __builtin_amdgcn_wmma_f32_16x16x32_f16(
                        false, a, false, bfrag, (short)0, acc, false, false);
            }
        }
        // C/D layout: lane ln -> N=o; VGPR r -> M = r + 8*hi => contiguous W addrs
        const int o = nbase + ln;
        const float bias = conv_b[o];
        float* obase = out + (((size_t)b * C_ + o) * H_ + h) * W_ + mbase + hi * 8;
        const float4 s0 = { acc[0] + bias, acc[1] + bias, acc[2] + bias, acc[3] + bias };
        const float4 s1 = { acc[4] + bias, acc[5] + bias, acc[6] + bias, acc[7] + bias };
        *reinterpret_cast<float4*>(obase)     = s0;
        *reinterpret_cast<float4*>(obase + 4) = s1;
    }
}

// ---------------------------------------------------------------------------
extern "C" void kernel_launch(void* const* d_in, const int* in_sizes, int n_in,
                              void* d_out, int out_size, void* d_ws, size_t ws_size,
                              hipStream_t stream) {
    (void)in_sizes; (void)n_in; (void)out_size; (void)ws_size;
    const float* x      = (const float*)d_in[0];
    const float* conv_w = (const float*)d_in[1];
    const float* conv_b = (const float*)d_in[2];
    const float* off_w  = (const float*)d_in[3];
    const float* off_b  = (const float*)d_in[4];
    const float* mask_w = (const float*)d_in[5];
    const float* mask_b = (const float*)d_in[6];
    float* out = (float*)d_out;

    // workspace: xt (NHWC fp32, 128 MB), packed conv weights, packed off/mask weights
    const size_t XT_BYTES = (size_t)B_ * H_ * W_ * C_ * sizeof(float);
    float*    xtw    = (float*)d_ws;
    _Float16* wt16   = (_Float16*)((char*)d_ws + XT_BYTES);
    _Float16* offw16 = (_Float16*)((char*)d_ws + XT_BYTES + (size_t)KT * C_ * C_ * 2);

    (void)hipFuncSetAttribute((const void*)deform_conv_main,
                              hipFuncAttributeMaxDynamicSharedMemorySize, LDS_TOTAL);

    transpose_nchw_nhwc<<<B_ * H_ * 16, 256, 0, stream>>>(x, xtw);
    pack_weights<<<(KT * C_ * C_ + 255) / 256, 256, 0, stream>>>(conv_w, wt16);
    pack_offw<<<(16 * OFFW_STRIDE + 255) / 256, 256, 0, stream>>>(off_w, mask_w, offw16);
    deform_conv_main<<<B_ * H_, 256, LDS_TOTAL, stream>>>(
        xtw, wt16, offw16, off_b, mask_b, conv_b, out);
}
